// NeuroSpikeSNN_53180285059360
// MI455X (gfx1250) — matrix operand order repaired and checked by reading, not compiled, not run on hardware
//
#include <hip/hip_runtime.h>

typedef _Float16 v16h __attribute__((ext_vector_type(16)));
typedef _Float16 v8h  __attribute__((ext_vector_type(8)));
typedef float    v8f  __attribute__((ext_vector_type(8)));

// ---------------------------------------------------------------------------
// WMMA fragment loaders (wave32, 16x16x32 f16), this code's contiguous k order k = 16h + e
// ---------------------------------------------------------------------------

// B fragment for C = A * W^T : B[k][n] = W[n][k], W row-major [N, ldw].
// Lane L holds col n = nbase + (L&15); K = kbase + (L<16 ? e : 16+e), e=0..15
// -> 16 contiguous halfs (one 32-byte load pair).
__device__ __forceinline__ v16h load_bfragT(const _Float16* __restrict__ W, int ldw,
                                            int nbase, int kbase, int lane) {
  int n = nbase + (lane & 15);
  int k = kbase + ((lane & 16) ? 16 : 0);
  return *(const v16h*)(W + (size_t)n * ldw + k);
}

// A fragment from row-major X[M, ldx].
// Lane L holds row m = mbase + (L&15); elements 0..15 -> K kb+{0..15} (or +16)
// -> two 16-byte loads.
__device__ __forceinline__ v16h load_afrag(const _Float16* __restrict__ X, int ldx,
                                           int mbase, int kbase, int lane) {
  int m  = mbase + (lane & 15);
  int k0 = kbase + ((lane & 16) ? 16 : 0);
  const _Float16* p = X + (size_t)m * ldx + k0;
  v8h lo = *(const v8h*)(p);
  v8h hi = *(const v8h*)(p + 8);
  v16h r;
#pragma unroll
  for (int i = 0; i < 8; ++i) { r[i] = lo[i]; r[8 + i] = hi[i]; }
  return r;
}

__device__ __forceinline__ float clamp01(float v) {
  return fminf(fmaxf(v, 0.0f), 1.0f);
}

// ---------------------------------------------------------------------------
// f32 -> f16 conversion
// ---------------------------------------------------------------------------
__global__ void k_cvt_f16(const float* __restrict__ in, _Float16* __restrict__ out, int n) {
  int i = blockIdx.x * blockDim.x + threadIdx.x;
  if (i < n) out[i] = (_Float16)in[i];
}

// ---------------------------------------------------------------------------
// Encoder: ench = LayerNorm(xh @ W_enc^T + b_enc), M=16384 K=384 N=512
// Block: 512 thr (16 waves) -> 64 rows x 512 cols.
// Wave w: M-tile (w>>2)*16, N-group (w&3)*128 (8 WMMA col-tiles).
// ---------------------------------------------------------------------------
__global__ __launch_bounds__(512) void k_enc(const _Float16* __restrict__ xh,
                                             const _Float16* __restrict__ Weh,
                                             const float* __restrict__ b_enc,
                                             const float* __restrict__ ln_g,
                                             const float* __restrict__ ln_b,
                                             _Float16* __restrict__ ench) {
  extern __shared__ char sm[];
  float* hbuf = (float*)sm;              // 64*512 f32 = 131072 B
  float* red  = (float*)(sm + 131072);   // 64*8
  float* red2 = red + 512;               // 64*8
  float* mu_s = red2 + 512;              // 64
  float* rs_s = mu_s + 64;               // 64

  const int KD = 384;
  int mbase = blockIdx.x * 64;
  int wave  = threadIdx.x >> 5;
  int lane  = threadIdx.x & 31;
  int mtile = (wave >> 2) * 16;
  int nwave = (wave & 3) * 128;

  v8f acc[8] = {};
  for (int kb = 0; kb < KD; kb += 32) {
    v16h a = load_afrag(xh, KD, mbase + mtile, kb, lane);
#pragma unroll
    for (int t = 0; t < 8; ++t) {
      v16h b = load_bfragT(Weh, KD, nwave + t * 16, kb, lane);
      acc[t] = __builtin_amdgcn_wmma_f32_16x16x32_f16(false, a, false, b,
                                                      (short)0, acc[t], false, false);
    }
  }

  int row0 = mtile + ((lane & 16) ? 8 : 0);
  int col  = lane & 15;
#pragma unroll
  for (int t = 0; t < 8; ++t) {
    int n = nwave + t * 16 + col;
    float bb = b_enc[n];
#pragma unroll
    for (int r = 0; r < 8; ++r) hbuf[(row0 + r) * 512 + n] = acc[t][r] + bb;
  }
  __syncthreads();

  {  // per-row mean/var: 8 partial threads per row (512 thr / 64 rows)
    int row  = threadIdx.x >> 3;
    int part = threadIdx.x & 7;
    float s = 0.f, s2 = 0.f;
    for (int c = part * 64; c < part * 64 + 64; ++c) {
      float v = hbuf[row * 512 + c]; s += v; s2 += v * v;
    }
    red[row * 8 + part] = s; red2[row * 8 + part] = s2;
  }
  __syncthreads();
  if (threadIdx.x < 64) {
    float s = 0.f, s2 = 0.f;
#pragma unroll
    for (int p = 0; p < 8; ++p) { s += red[threadIdx.x * 8 + p]; s2 += red2[threadIdx.x * 8 + p]; }
    float mu  = s * (1.0f / 512.0f);
    float var = s2 * (1.0f / 512.0f) - mu * mu;
    mu_s[threadIdx.x] = mu;
    rs_s[threadIdx.x] = rsqrtf(var + 1e-5f);
  }
  __syncthreads();
  for (int i = threadIdx.x; i < 64 * 512; i += 512) {
    int row = i >> 9, c = i & 511;
    float v = (hbuf[i] - mu_s[row]) * rs_s[row] * ln_g[c] + ln_b[c];
    ench[(size_t)(mbase + row) * 512 + c] = (_Float16)v;
  }
}

// ---------------------------------------------------------------------------
// cur1 = ench @ W1^T + b1 (loop-invariant input current), M=16384 K=512 N=512
// Block: 512 thr, 64 rows x 512 cols.
// ---------------------------------------------------------------------------
__global__ __launch_bounds__(512) void k_cur1(const _Float16* __restrict__ ench,
                                              const _Float16* __restrict__ W1h,
                                              const float* __restrict__ b1,
                                              float* __restrict__ cur1) {
  const int KD = 512;
  int mbase = blockIdx.x * 64;
  int wave  = threadIdx.x >> 5;
  int lane  = threadIdx.x & 31;
  int mtile = (wave >> 2) * 16;
  int nwave = (wave & 3) * 128;

  v8f acc[8] = {};
  for (int kb = 0; kb < KD; kb += 32) {
    v16h a = load_afrag(ench, KD, mbase + mtile, kb, lane);
#pragma unroll
    for (int t = 0; t < 8; ++t) {
      v16h b = load_bfragT(W1h, KD, nwave + t * 16, kb, lane);
      acc[t] = __builtin_amdgcn_wmma_f32_16x16x32_f16(false, a, false, b,
                                                      (short)0, acc[t], false, false);
    }
  }
  int row0 = mtile + ((lane & 16) ? 8 : 0);
  int col  = lane & 15;
#pragma unroll
  for (int t = 0; t < 8; ++t) {
    int n = nwave + t * 16 + col;
    float bb = b1[n];
#pragma unroll
    for (int r = 0; r < 8; ++r)
      cur1[(size_t)(mbase + row0 + r) * 512 + n] = acc[t][r] + bb;
  }
}

// ---------------------------------------------------------------------------
// Fused 10-step LIF recurrence + rate + decoder.
// Block: 512 thr (16 waves, 4 M-tiles x 4 N-groups) owns 64 batch rows.
// LDS: m1[64][512]f32 (128K) | s1[64][512]f16 (64K) | s2[64][256]f16 (32K)
//      | W3 staged [128][256]f16 (64K)  => 288 KB total.
// Decoder scratch (rate/d1/d2) aliases the dead m1 region after the loop.
// m2 / m3 / rate live in registers in the fixed WMMA C/D lane mapping.
// ---------------------------------------------------------------------------
__global__ __launch_bounds__(512) void k_steps(const float* __restrict__ cur1g,
                                               const _Float16* __restrict__ W2h,
                                               const _Float16* __restrict__ W3h,
                                               const float* __restrict__ beta1p,
                                               const float* __restrict__ b2p,
                                               const float* __restrict__ beta2p,
                                               const float* __restrict__ b3p,
                                               const float* __restrict__ beta3p,
                                               const float* __restrict__ Wd1,
                                               const float* __restrict__ bd1,
                                               const float* __restrict__ Wd2,
                                               const float* __restrict__ bd2,
                                               const float* __restrict__ Wd3,
                                               const float* __restrict__ bd3,
                                               float* __restrict__ out) {
  extern __shared__ char smem[];
  float*    m1    = (float*)smem;                          // 131072 B
  _Float16* s1    = (_Float16*)(smem + 131072);            //  65536 B
  _Float16* s2    = (_Float16*)(smem + 131072 + 65536);    //  32768 B
  _Float16* W3lds = (_Float16*)(smem + 131072 + 65536 + 32768); // 65536 B
  // decoder scratch aliases m1 (dead after the time loop)
  float* ratebuf = (float*)smem;        // 64*128 f32
  float* d1      = ratebuf + 64 * 128;  // 64*64
  float* d2      = d1 + 64 * 64;        // 64*32

  int mbase = blockIdx.x * 64;
  int tid   = threadIdx.x;
  int wave  = tid >> 5;
  int lane  = tid & 31;
  int mtile = (wave >> 2) * 16;
  int ngrp  = wave & 3;
  int row0  = mtile + ((lane & 16) ? 8 : 0);
  int col   = lane & 15;

  float beta1c = clamp01(beta1p[0]);
  float beta2c = clamp01(beta2p[0]);
  float beta3c = clamp01(beta3p[0]);

  // stage W3 (128x256 f16 = 64 KB) into LDS; init m1 = 0
  {
    const v8h* src = (const v8h*)W3h;       // 16-byte chunks
    v8h* dst = (v8h*)W3lds;
    for (int i = tid; i < (128 * 256) / 8; i += 512) dst[i] = src[i];
  }
  for (int i = tid; i < 64 * 512; i += 512) m1[i] = 0.0f;

  float m2reg[4][8];
  float m3reg[2][8];
  float rateg[2][8];
#pragma unroll
  for (int t = 0; t < 4; ++t)
#pragma unroll
    for (int r = 0; r < 8; ++r) m2reg[t][r] = 0.0f;
#pragma unroll
  for (int t = 0; t < 2; ++t)
#pragma unroll
    for (int r = 0; r < 8; ++r) { m3reg[t][r] = 0.0f; rateg[t][r] = 0.0f; }
  __syncthreads();

  for (int step = 0; step < 10; ++step) {
    // ---- LIF layer 1 (elementwise; cur1 stays L2-resident) ----
#pragma unroll 4
    for (int j = 0; j < 64; ++j) {
      int i = j * 512 + tid;
      int row = i >> 9, c = i & 511;
      float m = m1[i];
      float reset = (m > 1.0f) ? 1.0f : 0.0f;
      m = beta1c * m + cur1g[(size_t)(mbase + row) * 512 + c] - reset;
      m1[i] = m;
      s1[i] = (_Float16)((m > 1.0f) ? 1.0f : 0.0f);
    }
    __syncthreads();

    // ---- layer 2: cur2 = s1 @ W2^T  (M=64, K=512, N=256; wave -> 16x64) ----
    v8f acc2[4] = {};
    for (int kb = 0; kb < 512; kb += 32) {
      v16h a = load_afrag(s1, 512, mtile, kb, lane);
#pragma unroll
      for (int t = 0; t < 4; ++t) {
        v16h b = load_bfragT(W2h, 512, ngrp * 64 + t * 16, kb, lane);
        acc2[t] = __builtin_amdgcn_wmma_f32_16x16x32_f16(false, a, false, b,
                                                         (short)0, acc2[t], false, false);
      }
    }
#pragma unroll
    for (int t = 0; t < 4; ++t) {
      int n = ngrp * 64 + t * 16 + col;
      float bb = b2p[n];
#pragma unroll
      for (int r = 0; r < 8; ++r) {
        float m = m2reg[t][r];
        float reset = (m > 1.0f) ? 1.0f : 0.0f;
        m = beta2c * m + acc2[t][r] + bb - reset;
        m2reg[t][r] = m;
        s2[(row0 + r) * 256 + n] = (_Float16)((m > 1.0f) ? 1.0f : 0.0f);
      }
    }
    __syncthreads();

    // ---- layer 3: cur3 = s2 @ W3^T from LDS (M=64, K=256, N=128) ----
    v8f acc3[2] = {};
    for (int kb = 0; kb < 256; kb += 32) {
      v16h a = load_afrag(s2, 256, mtile, kb, lane);
#pragma unroll
      for (int t = 0; t < 2; ++t) {
        v16h b = load_bfragT(W3lds, 256, ngrp * 32 + t * 16, kb, lane);
        acc3[t] = __builtin_amdgcn_wmma_f32_16x16x32_f16(false, a, false, b,
                                                         (short)0, acc3[t], false, false);
      }
    }
#pragma unroll
    for (int t = 0; t < 2; ++t) {
      int n = ngrp * 32 + t * 16 + col;
      float bb = b3p[n];
#pragma unroll
      for (int r = 0; r < 8; ++r) {
        float m = m3reg[t][r];
        float reset = (m > 1.0f) ? 1.0f : 0.0f;
        m = beta3c * m + acc3[t][r] + bb - reset;
        m3reg[t][r] = m;
        rateg[t][r] += (m > 1.0f) ? 1.0f : 0.0f;
      }
    }
    __syncthreads();
  }

  // rate = spikes/10 -> LDS (aliases dead m1) for decoder
#pragma unroll
  for (int t = 0; t < 2; ++t) {
    int n = ngrp * 32 + t * 16 + col;
#pragma unroll
    for (int r = 0; r < 8; ++r)
      ratebuf[(row0 + r) * 128 + n] = rateg[t][r] * 0.1f;
  }
  __syncthreads();

  // ---- decoder (tiny: plain f32 FMA) ----
  for (int i = tid; i < 64 * 64; i += 512) {
    int row = i >> 6, c = i & 63;
    float s = bd1[c];
    const float* w = Wd1 + (size_t)c * 128;
    for (int k = 0; k < 128; ++k) s += ratebuf[row * 128 + k] * w[k];
    d1[i] = fmaxf(s, 0.0f);
  }
  __syncthreads();
  for (int i = tid; i < 64 * 32; i += 512) {
    int row = i >> 5, c = i & 31;
    float s = bd2[c];
    const float* w = Wd2 + (size_t)c * 64;
    for (int k = 0; k < 64; ++k) s += d1[row * 64 + k] * w[k];
    d2[i] = fmaxf(s, 0.0f);
  }
  __syncthreads();
  for (int i = tid; i < 64 * 24; i += 512) {
    int row = i / 24, c = i % 24;
    float s = bd3[c];
    const float* w = Wd3 + (size_t)c * 32;
    for (int k = 0; k < 32; ++k) s += d2[row * 32 + k] * w[k];
    out[(size_t)(mbase + row) * 24 + c] = s;
  }
}

// ---------------------------------------------------------------------------
// Host launcher
// ---------------------------------------------------------------------------
extern "C" void kernel_launch(void* const* d_in, const int* in_sizes, int n_in,
                              void* d_out, int out_size, void* d_ws, size_t ws_size,
                              hipStream_t stream) {
  (void)in_sizes; (void)n_in; (void)out_size; (void)ws_size;
  const float* x     = (const float*)d_in[0];
  const float* W_enc = (const float*)d_in[1];
  const float* b_enc = (const float*)d_in[2];
  const float* ln_g  = (const float*)d_in[3];
  const float* ln_b  = (const float*)d_in[4];
  const float* W1    = (const float*)d_in[5];
  const float* b1    = (const float*)d_in[6];
  const float* beta1 = (const float*)d_in[7];
  const float* W2    = (const float*)d_in[8];
  const float* b2    = (const float*)d_in[9];
  const float* beta2 = (const float*)d_in[10];
  const float* W3    = (const float*)d_in[11];
  const float* b3    = (const float*)d_in[12];
  const float* beta3 = (const float*)d_in[13];
  const float* Wd1   = (const float*)d_in[14];
  const float* bd1   = (const float*)d_in[15];
  const float* Wd2   = (const float*)d_in[16];
  const float* bd2   = (const float*)d_in[17];
  const float* Wd3   = (const float*)d_in[18];
  const float* bd3   = (const float*)d_in[19];
  float* out = (float*)d_out;

  const int B = 16384, D = 384, H0 = 512, H1 = 256, H2 = 128;

  char* ws = (char*)d_ws;
  size_t off = 0;
  auto take = [&](size_t bytes) -> char* {
    char* p = ws + off;
    off = (off + bytes + 255) & ~(size_t)255;
    return p;
  };
  _Float16* xh   = (_Float16*)take((size_t)B * D * 2);
  _Float16* Weh  = (_Float16*)take((size_t)H0 * D * 2);
  _Float16* W1h  = (_Float16*)take((size_t)H0 * H0 * 2);
  _Float16* W2h  = (_Float16*)take((size_t)H1 * H0 * 2);
  _Float16* W3h  = (_Float16*)take((size_t)H2 * H1 * 2);
  _Float16* ench = (_Float16*)take((size_t)B * H0 * 2);
  float*    cur1 = (float*)take((size_t)B * H0 * 4);

  auto cvt = [&](const float* src, _Float16* dst, int n) {
    k_cvt_f16<<<dim3((n + 255) / 256), dim3(256), 0, stream>>>(src, dst, n);
  };
  cvt(x,     xh,  B * D);
  cvt(W_enc, Weh, H0 * D);
  cvt(W1,    W1h, H0 * H0);
  cvt(W2,    W2h, H1 * H0);
  cvt(W3,    W3h, H2 * H1);

  size_t smem_enc = 131072 + 2048 + 2048 + 256 + 256;  // hbuf + red + red2 + mu + rs
  k_enc<<<dim3(B / 64), dim3(512), smem_enc, stream>>>(xh, Weh, b_enc, ln_g, ln_b, ench);
  k_cur1<<<dim3(B / 64), dim3(512), 0, stream>>>(ench, W1h, b1, cur1);

  size_t smem_steps = 131072 + 65536 + 32768 + 65536;  // m1 + s1 + s2 + W3  = 288 KB
  k_steps<<<dim3(B / 64), dim3(512), smem_steps, stream>>>(
      cur1, W2h, W3h, beta1, b2, beta2, b3, beta3,
      Wd1, bd1, Wd2, bd2, Wd3, bd3, out);
}
